// BiLstm_CRF_13340168421804
// MI455X (gfx1250) — compile-verified, matrix operand order checked by reading, not run on hardware
//
#include <hip/hip_runtime.h>
#include <hip/hip_bf16.h>

// ---------------------------------------------------------------------------
// BiLSTM-CRF for MI455X (gfx1250, wave32, WMMA + async-LDS staging).
//   Phase 1: embedding gather + bf16 casts
//   Phase 2: WMMA bf16 GEMM  Zx = x @ Wih.T + (bih + bhh)  (async-LDS tiled)
//   Phase 3: sequential LSTM recurrence (1 WGP per direction, LDS-resident h/c)
//   Phase 4: WMMA bf16 GEMM  feats = lstm_out @ W_out.T + b_out (direct)
//   Phase 5: Viterbi forward pass + backtrace (single WGP, CRF in LDS)
// ---------------------------------------------------------------------------

#define S_LEN 2048
#define E_DIM 512
#define H_DIM 512
#define G_DIM (4 * H_DIM)   // 2048 gate pre-activations
#define HD_DIM (2 * H_DIM)  // 1024 concat hidden
#define T_TAGS 48
#define NEG_INF (-10000.0f)

// Macro-tile config for the async-LDS tiled GEMM
#define BM 64
#define BN 64
#define BK 32

typedef __attribute__((ext_vector_type(16))) __bf16 v16bf;
typedef __attribute__((ext_vector_type(8)))  float  v8f;

__device__ __forceinline__ unsigned short f32_to_bf16_bits(float f) {
    union { float f; unsigned int u; } v;
    v.f = f;
    unsigned int u = v.u;
    unsigned int r = u + 0x7FFFu + ((u >> 16) & 1u);  // round-to-nearest-even
    return (unsigned short)(r >> 16);
}

__device__ __forceinline__ float sigmoidf_dev(float x) {
    return 1.0f / (1.0f + __expf(-x));
}

// Low 32 bits of a generic shared-memory address == LDS byte offset
__device__ __forceinline__ unsigned lds_offset_of(const void* p) {
    return (unsigned)(size_t)p;
}

// ---------------------------------------------------------------------------
// Kernel 1: embedding gather + fp32 -> bf16 cast
// ---------------------------------------------------------------------------
__global__ void embed_cast_kernel(const float* __restrict__ emb,
                                  const int* __restrict__ sentence,
                                  unsigned short* __restrict__ xbf) {
    int idx = blockIdx.x * blockDim.x + threadIdx.x;
    if (idx >= S_LEN * E_DIM) return;
    int s = idx >> 9;              // / E_DIM
    int e = idx & (E_DIM - 1);
    int tok = sentence[s];
    xbf[idx] = f32_to_bf16_bits(emb[(size_t)tok * E_DIM + e]);
}

// ---------------------------------------------------------------------------
// Kernel 2: generic fp32 -> bf16 cast (weights)
// ---------------------------------------------------------------------------
__global__ void cast_bf16_kernel(const float* __restrict__ src,
                                 unsigned short* __restrict__ dst, int n) {
    int i = blockIdx.x * blockDim.x + threadIdx.x;
    if (i < n) dst[i] = f32_to_bf16_bits(src[i]);
}

// ---------------------------------------------------------------------------
// Kernel 3a: async-LDS tiled WMMA GEMM:  C[M,N] = A[M,K] * B[N,K]^T + b1 + b2
// Block = 512 threads (16 waves) -> one 64x64 C macro-tile (4x4 WMMA tiles).
// Operand panels (64x32 bf16 each) staged into LDS with
// global_load_async_to_lds_b128 (ASYNCcnt), double-buffered across K chunks.
// Dimensions must divide BM/BN/BK exactly (true for the input projections).
// ---------------------------------------------------------------------------
__global__ __launch_bounds__(512)
void wmma_gemm_tiled_kernel(const unsigned short* __restrict__ A, int lda,
                            const unsigned short* __restrict__ B, int ldb,
                            const float* __restrict__ bias1,
                            const float* __restrict__ bias2,
                            float* __restrict__ C, int ldc, int K) {
    __shared__ unsigned short __attribute__((aligned(32))) smA[2][BM * BK];
    __shared__ unsigned short __attribute__((aligned(32))) smB[2][BN * BK];

    const int tid  = threadIdx.x;          // 0..511
    const int wave = tid >> 5;             // 0..15
    const int lane = tid & 31;
    const int wm   = wave >> 2;            // 0..3 (M sub-tile)
    const int wn   = wave & 3;             // 0..3 (N sub-tile)
    const int mBase = blockIdx.y * BM;
    const int nBase = blockIdx.x * BN;

    // Staging roles: threads 0..255 stage A panel, 256..511 stage B panel.
    // Each thread moves one 16B (b128) chunk per K-chunk.
    const bool doB  = tid >= 256;
    const int  sidx = doB ? (tid - 256) : tid;   // 0..255
    const int  srow = sidx >> 2;                 // 0..63
    const int  scol = (sidx & 3) * 8;            // bf16 elements: 0,8,16,24
    const unsigned short* __restrict__ gsrc =
        (doB ? B + (size_t)(nBase + srow) * ldb
             : A + (size_t)(mBase + srow) * lda) + scol;
    const unsigned ldsBase =
        lds_offset_of(doB ? &smB[0][srow * BK + scol]
                          : &smA[0][srow * BK + scol]);
    const unsigned bufStride = (unsigned)(BM * BK * sizeof(unsigned short)); // 4KB

    // Fragment offsets within a panel (32B-contiguous per lane).
    const int kHalf = (lane >> 4) << 4;   // 0 or 16
    const int fOffA = (wm * 16 + (lane & 15)) * BK + kHalf;
    const int fOffB = (wn * 16 + (lane & 15)) * BK + kHalf;

    v8f c = {0.f, 0.f, 0.f, 0.f, 0.f, 0.f, 0.f, 0.f};

    const int nChunks = K / BK;

    // Prologue: stage chunk 0 into buffer 0.
    asm volatile("global_load_async_to_lds_b128 %0, %1, off"
                 :: "v"(ldsBase), "v"(gsrc) : "memory");

    for (int kc = 0; kc < nChunks; ++kc) {
        const int cur = kc & 1;
        asm volatile("s_wait_asynccnt 0" ::: "memory");
        __syncthreads();   // buffer[cur] fully populated across the block

        // Issue async loads for the next chunk into the other buffer.
        if (kc + 1 < nChunks) {
            const unsigned short* gnext = gsrc + (size_t)(kc + 1) * BK;
            const unsigned ldsNext = ldsBase + (cur ^ 1) * bufStride;
            asm volatile("global_load_async_to_lds_b128 %0, %1, off"
                         :: "v"(ldsNext), "v"(gnext) : "memory");
        }

        // Compute from buffer[cur]: LDS fragment reads + WMMA.
        v16bf a = *reinterpret_cast<const v16bf*>(&smA[cur][fOffA]);
        v16bf b = *reinterpret_cast<const v16bf*>(&smB[cur][fOffB]);
        c = __builtin_amdgcn_wmma_f32_16x16x32_bf16(
                /*neg_a=*/false, a, /*neg_b=*/false, b,
                /*c_mod=*/(short)0, c, /*reuse_a=*/false, /*reuse_b=*/false);

        __syncthreads();   // all reads of buffer[cur] done before its reuse
    }

    // Epilogue: C/D f32 layout (VGPR r, lanes 0-15 -> M=r, 16-31 -> M=8+r).
    const int n  = nBase + wn * 16 + (lane & 15);
    float bv = bias1[n] + (bias2 ? bias2[n] : 0.0f);
    const int mOff = (lane >> 4) * 8;
#pragma unroll
    for (int r = 0; r < 8; ++r) {
        int m = mBase + wm * 16 + mOff + r;
        C[(size_t)m * ldc + n] = c[r] + bv;
    }
}

// ---------------------------------------------------------------------------
// Kernel 3b: direct per-wave WMMA GEMM (used for the small N=48 projection).
// ---------------------------------------------------------------------------
__global__ void wmma_gemm_bias_kernel(const unsigned short* __restrict__ A, int lda,
                                      const unsigned short* __restrict__ B, int ldb,
                                      const float* __restrict__ bias1,
                                      const float* __restrict__ bias2,
                                      float* __restrict__ C, int ldc,
                                      int N, int K) {
    const int wave  = threadIdx.x >> 5;
    const int lane  = threadIdx.x & 31;
    const int tileN = blockIdx.x * 4 + wave;
    const int tileM = blockIdx.y;
    const int tilesN = N >> 4;
    if (tileN >= tilesN) return;   // uniform per wave: EXEC stays all-1s

    const int mBase = tileM << 4;
    const int nBase = tileN << 4;
    const int rowA  = mBase + (lane & 15);
    const int rowB  = nBase + (lane & 15);
    const int kHalf = (lane >> 4) << 4;

    v8f c = {0.f, 0.f, 0.f, 0.f, 0.f, 0.f, 0.f, 0.f};

    const unsigned short* aPtr = A + (size_t)rowA * lda + kHalf;
    const unsigned short* bPtr = B + (size_t)rowB * ldb + kHalf;
    for (int k0 = 0; k0 < K; k0 += 32) {
        v16bf a = *reinterpret_cast<const v16bf*>(aPtr + k0);
        v16bf b = *reinterpret_cast<const v16bf*>(bPtr + k0);
        c = __builtin_amdgcn_wmma_f32_16x16x32_bf16(
                false, a, false, b, (short)0, c, false, false);
    }

    const int n  = nBase + (lane & 15);
    float bv = bias1[n] + (bias2 ? bias2[n] : 0.0f);
    const int mOff = (lane >> 4) * 8;
#pragma unroll
    for (int r = 0; r < 8; ++r) {
        int m = mBase + mOff + r;
        C[(size_t)m * ldc + n] = c[r] + bv;
    }
}

// ---------------------------------------------------------------------------
// Kernel 4: sequential LSTM recurrence. gridDim.x == 2 (dir), block = 1024.
// ---------------------------------------------------------------------------
__global__ __launch_bounds__(1024)
void lstm_recurrence_kernel(const float* __restrict__ Zx_f,
                            const float* __restrict__ Zx_b,
                            const float* __restrict__ Whh_f,
                            const float* __restrict__ Whh_b,
                            const float* __restrict__ h0,
                            const float* __restrict__ c0,
                            unsigned short* __restrict__ lstm_bf) {
    const int dir = blockIdx.x;            // 0 = forward, 1 = backward
    const float* __restrict__ Zx  = dir ? Zx_b  : Zx_f;
    const float* __restrict__ Whh = dir ? Whh_b : Whh_f;

    __shared__ float h_s[H_DIM];
    __shared__ float c_s[H_DIM];
    __shared__ float z_s[G_DIM];

    const int tid = threadIdx.x;           // 0..1023
    if (tid < H_DIM) {
        h_s[tid] = h0[dir * H_DIM + tid];
        c_s[tid] = c0[dir * H_DIM + tid];
    }
    __syncthreads();

    for (int step = 0; step < S_LEN; ++step) {
        const int t = dir ? (S_LEN - 1 - step) : step;
        const float* __restrict__ zrow = Zx + (size_t)t * G_DIM;
        if (step + 1 < S_LEN) {
            const int tn = dir ? (S_LEN - 2 - step) : (step + 1);
            __builtin_prefetch(Zx + (size_t)tn * G_DIM + (tid & 511), 0, 0);
        }

#pragma unroll
        for (int jj = 0; jj < 2; ++jj) {
            const int j = tid + jj * 1024;
            const float4* __restrict__ w =
                reinterpret_cast<const float4*>(Whh + (size_t)j * H_DIM);
            float acc = 0.0f;
#pragma unroll 4
            for (int k4 = 0; k4 < H_DIM / 4; ++k4) {
                float4 wv = w[k4];
                acc += wv.x * h_s[k4 * 4 + 0];
                acc += wv.y * h_s[k4 * 4 + 1];
                acc += wv.z * h_s[k4 * 4 + 2];
                acc += wv.w * h_s[k4 * 4 + 3];
            }
            z_s[j] = acc + zrow[j];
        }
        __syncthreads();

        if (tid < H_DIM) {
            float iv = sigmoidf_dev(z_s[tid]);
            float fv = sigmoidf_dev(z_s[H_DIM + tid]);
            float gv = tanhf(z_s[2 * H_DIM + tid]);
            float ov = sigmoidf_dev(z_s[3 * H_DIM + tid]);
            float cn = fv * c_s[tid] + iv * gv;
            float hn = ov * tanhf(cn);
            c_s[tid] = cn;
            h_s[tid] = hn;
            lstm_bf[(size_t)t * HD_DIM + dir * H_DIM + tid] = f32_to_bf16_bits(hn);
        }
        __syncthreads();
    }
}

// ---------------------------------------------------------------------------
// Kernel 5: Viterbi (single workgroup of 64; lanes 0..47 = "to" states)
// ---------------------------------------------------------------------------
__global__ void viterbi_kernel(const float* __restrict__ feats,
                               const float* __restrict__ crf,
                               int* __restrict__ steps,
                               float* __restrict__ out) {
    __shared__ float crf_s[T_TAGS * T_TAGS];
    __shared__ float sp[T_TAGS];
    __shared__ float nsp[T_TAGS];

    const int tid = threadIdx.x;
    for (int i = tid; i < T_TAGS * T_TAGS; i += blockDim.x) crf_s[i] = crf[i];
    if (tid < T_TAGS) sp[tid] = (tid == T_TAGS - 2) ? 0.0f : NEG_INF;  // START
    __syncthreads();

    for (int t = 0; t < S_LEN; ++t) {
        if (tid < T_TAGS) {
            float best = -3.4e38f;
            int   bi   = 0;
            for (int from = 0; from < T_TAGS; ++from) {
                float s = sp[from] + crf_s[from * T_TAGS + tid];
                if (s > best) { best = s; bi = from; }
            }
            nsp[tid] = best + feats[t * T_TAGS + tid];
            steps[t * T_TAGS + tid] = bi;
        }
        __syncthreads();
        if (tid < T_TAGS) sp[tid] = nsp[tid];
        __syncthreads();
    }

    if (tid == 0) {
        float best = -3.4e38f;
        int   bi   = 0;
        for (int to = 0; to < T_TAGS; ++to) {
            float s = sp[to] + crf_s[to * T_TAGS + (T_TAGS - 1)];  // -> END
            if (s > best) { best = s; bi = to; }
        }
        out[S_LEN] = best;              // path score (output #2)
        int lbl = bi;
        for (int t = S_LEN - 1; t >= 0; --t) {
            out[t] = (float)lbl;        // labels as floats (output #1)
            lbl = steps[t * T_TAGS + lbl];
        }
    }
}

// ---------------------------------------------------------------------------
// Host launcher
// ---------------------------------------------------------------------------
extern "C" void kernel_launch(void* const* d_in, const int* in_sizes, int n_in,
                              void* d_out, int out_size, void* d_ws, size_t ws_size,
                              hipStream_t stream) {
    (void)in_sizes; (void)n_in; (void)out_size; (void)ws_size;

    const float* emb    = (const float*)d_in[0];
    const float* Wih_f  = (const float*)d_in[1];
    const float* Whh_f  = (const float*)d_in[2];
    const float* bih_f  = (const float*)d_in[3];
    const float* bhh_f  = (const float*)d_in[4];
    const float* Wih_b  = (const float*)d_in[5];
    const float* Whh_b  = (const float*)d_in[6];
    const float* bih_b  = (const float*)d_in[7];
    const float* bhh_b  = (const float*)d_in[8];
    const float* W_out  = (const float*)d_in[9];
    const float* b_out  = (const float*)d_in[10];
    const float* crf    = (const float*)d_in[11];
    const float* h0     = (const float*)d_in[12];
    const float* c0     = (const float*)d_in[13];
    const int*   sent   = (const int*)d_in[14];
    float*       out    = (float*)d_out;

    // ---- workspace layout (256B aligned slabs) ----
    char* ws = (char*)d_ws;
    size_t off = 0;
    auto take = [&](size_t bytes) -> char* {
        char* p = ws + off;
        off += (bytes + 255) & ~(size_t)255;
        return p;
    };
    unsigned short* xbf     = (unsigned short*)take((size_t)S_LEN * E_DIM * 2);
    unsigned short* wihf_bf = (unsigned short*)take((size_t)G_DIM * E_DIM * 2);
    unsigned short* wihb_bf = (unsigned short*)take((size_t)G_DIM * E_DIM * 2);
    unsigned short* wout_bf = (unsigned short*)take((size_t)T_TAGS * HD_DIM * 2);
    float*          zxf     = (float*)take((size_t)S_LEN * G_DIM * 4);
    float*          zxb     = (float*)take((size_t)S_LEN * G_DIM * 4);
    unsigned short* lstm_bf = (unsigned short*)take((size_t)S_LEN * HD_DIM * 2);
    float*          feats   = (float*)take((size_t)S_LEN * T_TAGS * 4);
    int*            steps   = (int*)take((size_t)S_LEN * T_TAGS * 4);

    // Phase 1: gather + casts
    {
        int n = S_LEN * E_DIM;
        embed_cast_kernel<<<(n + 255) / 256, 256, 0, stream>>>(emb, sent, xbf);
    }
    {
        int n = G_DIM * E_DIM;
        cast_bf16_kernel<<<(n + 255) / 256, 256, 0, stream>>>(Wih_f, wihf_bf, n);
        cast_bf16_kernel<<<(n + 255) / 256, 256, 0, stream>>>(Wih_b, wihb_bf, n);
        int nw = T_TAGS * HD_DIM;
        cast_bf16_kernel<<<(nw + 255) / 256, 256, 0, stream>>>(W_out, wout_bf, nw);
    }

    // Phase 2: input projections via async-LDS tiled WMMA (M=N=2048, K=512)
    {
        dim3 grid(G_DIM / BN, S_LEN / BM);   // (32, 32)
        dim3 block(512);
        wmma_gemm_tiled_kernel<<<grid, block, 0, stream>>>(
            xbf, E_DIM, wihf_bf, E_DIM, bih_f, bhh_f, zxf, G_DIM, E_DIM);
        wmma_gemm_tiled_kernel<<<grid, block, 0, stream>>>(
            xbf, E_DIM, wihb_bf, E_DIM, bih_b, bhh_b, zxb, G_DIM, E_DIM);
    }

    // Phase 3: sequential recurrence (one WGP per direction)
    lstm_recurrence_kernel<<<2, 1024, 0, stream>>>(
        zxf, zxb, Whh_f, Whh_b, h0, c0, lstm_bf);

    // Phase 4: output projection via direct WMMA (M=2048, N=48, K=1024)
    {
        dim3 grid((T_TAGS / 16 + 3) / 4, S_LEN / 16);
        dim3 block(128);
        wmma_gemm_bias_kernel<<<grid, block, 0, stream>>>(
            lstm_bf, HD_DIM, wout_bf, HD_DIM, b_out, nullptr, feats, T_TAGS,
            T_TAGS, HD_DIM);
    }

    // Phase 5: Viterbi decode
    viterbi_kernel<<<1, 64, 0, stream>>>(feats, crf, steps, out);
}
